// TimeLSTM_v3_4252017623221
// MI455X (gfx1250) — matrix operand order repaired and checked by reading, not compiled, not run on hardware
//
#include <hip/hip_runtime.h>

#define B_ 128
#define N_ 512
#define X_ 128
#define H_ 256

typedef __attribute__((ext_vector_type(16))) __bf16 v16bf;
typedef __attribute__((ext_vector_type(8)))  __bf16 v8bf;
typedef __attribute__((ext_vector_type(8)))  float  v8f;

__device__ __forceinline__ unsigned short f2bf(float f) {
    unsigned u = __float_as_uint(f);
    u += 0x7FFFu + ((u >> 16) & 1u);            // round-to-nearest-even
    return (unsigned short)(u >> 16);
}
__device__ __forceinline__ unsigned pack2bf(float a, float b) {
    return (unsigned)f2bf(a) | ((unsigned)f2bf(b) << 16);
}
__device__ __forceinline__ float bf2f(unsigned short u) {
    return __uint_as_float((unsigned)u << 16);
}
__device__ __forceinline__ float bf2f_lo(unsigned w) { return __uint_as_float(w << 16); }
__device__ __forceinline__ float bf2f_hi(unsigned w) { return __uint_as_float(w & 0xffff0000u); }

__device__ __forceinline__ float sigm(float x) { return 1.0f / (1.0f + __expf(-x)); }

__global__ void __launch_bounds__(256) cvt_bf16(const float* __restrict__ src,
                                                unsigned short* __restrict__ dst, int n) {
    int i = blockIdx.x * blockDim.x + threadIdx.x;
    if (i < n) dst[i] = f2bf(src[i]);
}

// A-fragment (16x32 bf16) from LDS, contiguous K order matching B loads:
// lane L: row m=L%16; K = kt*32 + klo .. +15, klo = 16*(L>=16)
__device__ __forceinline__ v16bf ldsA(const unsigned short* S, int ldk, int lane, int kt) {
    int m = lane & 15;
    int klo = kt * 32 + ((lane >> 4) << 4);
    v8bf lo = *(const v8bf*)(S + m * ldk + klo);
    v8bf hi = *(const v8bf*)(S + m * ldk + klo + 8);
    return __builtin_shufflevector(lo, hi, 0, 1, 2, 3, 4, 5, 6, 7,
                                   8, 9, 10, 11, 12, 13, 14, 15);
}

struct ColParams {
    const float *ig_w_c, *ig_b, *fg_w_c, *fg_b, *in_b,
                *og_w_cn, *og_b, *og_w_t,
                *tg1_w_t, *tg1_b, *tg2_w_t, *tg2_b;
};

// ---------------------------------------------------------------------------
// Precompute pass: for every (n, batch-tile) compute x_t @ Wx^T for all 6
// gates, store bf16 in transposed tile layout pre[n][wg][g][col 0..255][row
// 0..15] so the scan can init its WMMA accumulator with one b128 load.
// ---------------------------------------------------------------------------
__global__ void __launch_bounds__(256) xgemm_pre(
    const float* __restrict__ x,
    const unsigned short* __restrict__ wx,   // 6 x [H][X] bf16
    unsigned short* __restrict__ pre)
{
    __shared__ unsigned short sX[16 * X_];
    const int tid  = threadIdx.x;
    const int lane = tid & 31;
    const int wave = tid >> 5;
    const int bn = blockIdx.x;        // timestep
    const int bw = blockIdx.y;        // batch tile (16 rows)
    const int rowbase = bw * 16;

    { // stage x[rowbase.., bn, :] -> bf16
        int xr = tid >> 4, xc = (tid & 15) << 3;
        const float* xp = x + ((size_t)(rowbase + xr) * N_ + bn) * X_ + xc;
        unsigned short* dp = sX + xr * X_ + xc;
        #pragma unroll
        for (int q = 0; q < 8; ++q) dp[q] = f2bf(xp[q]);
    }
    __syncthreads();

    for (int task = wave; task < 96; task += 8) {
        const int g = task >> 4;
        const int c = task & 15;
        const int j  = c * 16 + (lane & 15);
        const int kb = (lane < 16) ? 0 : 16;
        v8f acc = {};
        const unsigned short* Wxg = wx + (size_t)g * H_ * X_ + (size_t)j * X_ + kb;
        #pragma unroll
        for (int kt = 0; kt < X_ / 32; ++kt) {
            v16bf b = *(const v16bf*)(Wxg + kt * 32);
            v16bf a = ldsA(sX, X_, lane, kt);
            acc = __builtin_amdgcn_wmma_f32_16x16x32_bf16(
                    false, a, false, b, (short)0, acc, false, false);
        }
        const int mb  = (lane < 16) ? 0 : 8;
        const int col = c * 16 + (lane & 15);
        unsigned short* dp = pre + ((((size_t)bn * 8 + bw) * 6 + g) * 256 + col) * 16 + mb;
        uint4 w;
        w.x = pack2bf(acc[0], acc[1]);
        w.y = pack2bf(acc[2], acc[3]);
        w.z = pack2bf(acc[4], acc[5]);
        w.w = pack2bf(acc[6], acc[7]);
        *(uint4*)dp = w;
    }
}

// ---------------------------------------------------------------------------
// Sequential scan. PRE=true: x-side GEMMs precomputed (C-init from `pre`);
// PRE=false: fused x-side GEMMs each step (no big workspace needed).
// Dynamic LDS layout (bytes):
//   sA   [16][H_] bf16 :      0 ..   8192
//   sX   [16][X_] bf16 :   8192 ..  12288   (fused path only)
//   gbuf [6][16][H_] f32:  12288 .. 110592  (PRE uses only gates 0..3)
//   pl   [12][H_] f32  : 110592 .. 122880
// ---------------------------------------------------------------------------
#define LDS_BYTES 122880
#define PRE_NSTRIDE ((size_t)8 * 6 * 256 * 16)   // ushorts per timestep

template <bool PRE>
__global__ void __launch_bounds__(256) timelstm_scan(
    const float* __restrict__ x, const float* __restrict__ t,
    const unsigned short* __restrict__ wh,   // 4 x [H][H] bf16: ig,fg,in,og
    const unsigned short* __restrict__ wx,   // 6 x [H][X] bf16
    const unsigned short* __restrict__ pre,  // precomputed x-side (PRE only)
    ColParams cp, float* __restrict__ out)
{
    extern __shared__ char smem[];
    unsigned short* sA = (unsigned short*)smem;            // h state, bf16
    unsigned short* sX = (unsigned short*)(smem + 8192);   // x tile, bf16
    float* gbuf = (float*)(smem + 12288);                  // gate pre-activations
    float* pl   = (float*)(smem + 110592);                 // per-column params

    const int tid  = threadIdx.x;
    const int lane = tid & 31;
    const int wave = tid >> 5;
    const int rowbase = blockIdx.x * 16;

    { // stage per-column params into LDS once
        const float* ps[12] = {cp.ig_w_c, cp.ig_b, cp.fg_w_c, cp.fg_b, cp.in_b,
                               cp.og_w_cn, cp.og_b, cp.og_w_t,
                               cp.tg1_w_t, cp.tg1_b, cp.tg2_w_t, cp.tg2_b};
        for (int i = tid; i < 12 * H_; i += 256) pl[i] = ps[i >> 8][i & 255];
    }
    for (int i = tid; i < 16 * H_; i += 256) sA[i] = 0;   // h0 = 0

    const int r  = tid >> 4;          // elementwise row (0..15)
    const int cb = (tid & 15) << 4;   // elementwise col base (16 cols/thread)
    float cmreg[16], hreg[16];
    #pragma unroll
    for (int q = 0; q < 16; ++q) { cmreg[q] = 0.f; hreg[q] = 0.f; }

    const int NTASK = PRE ? 64 : 96;

    for (int n = 0; n < N_; ++n) {
        if constexpr (!PRE) { // stage x[rowbase.., n, :] -> bf16 sX
            int xr = tid >> 4, xc = (tid & 15) << 3;
            const float* xp = x + ((size_t)(rowbase + xr) * N_ + n) * X_ + xc;
            unsigned short* dp = sX + xr * X_ + xc;
            #pragma unroll
            for (int q = 0; q < 8; ++q) dp[q] = f2bf(xp[q]);
        }
        __syncthreads();  // sA from previous step (and sX) safe to read

        for (int task = wave; task < NTASK; task += 8) {
            const int g = task >> 4;              // PRE: 0..3, fused: 0..5
            const int c = task & 15;
            const int j  = c * 16 + (lane & 15);  // weight row = output column
            const int kb = (lane < 16) ? 0 : 16;  // B-fragment K half
            const int mb  = (lane < 16) ? 0 : 8;
            const int col = c * 16 + (lane & 15);
            v8f acc;
            if constexpr (PRE) {
                // init accumulator with precomputed x-side preactivation
                const unsigned short* cp8 =
                    pre + ((((size_t)n * 8 + blockIdx.x) * 6 + g) * 256 + col) * 16 + mb;
                uint4 cw = *(const uint4*)cp8;
                acc[0] = bf2f_lo(cw.x); acc[1] = bf2f_hi(cw.x);
                acc[2] = bf2f_lo(cw.y); acc[3] = bf2f_hi(cw.y);
                acc[4] = bf2f_lo(cw.z); acc[5] = bf2f_hi(cw.z);
                acc[6] = bf2f_lo(cw.w); acc[7] = bf2f_hi(cw.w);
                if (n + 1 < N_)  // pull next step's tile toward L2/L0
                    __builtin_prefetch(cp8 + PRE_NSTRIDE, 0, 0);
            } else {
                acc = v8f{};
                const unsigned short* Wxg = wx + (size_t)g * H_ * X_ + (size_t)j * X_ + kb;
                #pragma unroll
                for (int kt = 0; kt < X_ / 32; ++kt) {
                    v16bf b = *(const v16bf*)(Wxg + kt * 32);
                    v16bf a = ldsA(sX, X_, lane, kt);
                    acc = __builtin_amdgcn_wmma_f32_16x16x32_bf16(
                            false, a, false, b, (short)0, acc, false, false);
                }
            }
            if (g < 4) { // recurrent side: acc += h @ Wh[g]^T  (K = 256)
                const unsigned short* Whg = wh + (size_t)g * H_ * H_ + (size_t)j * H_ + kb;
                #pragma unroll
                for (int kt = 0; kt < H_ / 32; ++kt) {
                    v16bf b = *(const v16bf*)(Whg + kt * 32);
                    v16bf a = ldsA(sA, H_, lane, kt);
                    acc = __builtin_amdgcn_wmma_f32_16x16x32_bf16(
                            false, a, false, b, (short)0, acc, false, false);
                }
            }
            // C/D layout: elem i -> M = i + 8*(lane>=16), N = lane%16
            #pragma unroll
            for (int i = 0; i < 8; ++i)
                gbuf[(g * 16 + mb + i) * H_ + col] = acc[i];
        }
        __syncthreads();  // gbuf ready; sA/sX reads complete

        { // elementwise gate math; cm lives in registers
            float tval = t[(size_t)(rowbase + r) * N_ + n];
            const unsigned short* tg = PRE ?
                pre + (((size_t)n * 8 + blockIdx.x) * 6) * 256 * 16 : nullptr;
            #pragma unroll
            for (int q = 0; q < 16; ++q) {
                int c = cb + q;
                float igp = gbuf[(0 * 16 + r) * H_ + c] + pl[1 * H_ + c];
                float fgp = gbuf[(1 * 16 + r) * H_ + c] + pl[3 * H_ + c];
                float inp = gbuf[(2 * 16 + r) * H_ + c] + pl[4 * H_ + c];
                float ogp = gbuf[(3 * 16 + r) * H_ + c];
                float t1p, t2p;
                if constexpr (PRE) {
                    t1p = bf2f(tg[(4 * 256 + c) * 16 + r]);
                    t2p = bf2f(tg[(5 * 256 + c) * 16 + r]);
                } else {
                    t1p = gbuf[(4 * 16 + r) * H_ + c];
                    t2p = gbuf[(5 * 16 + r) * H_ + c];
                }
                float cm = cmreg[q];
                float ig = sigm(pl[0 * H_ + c] * cm + igp);
                float fg = sigm(pl[2 * H_ + c] * cm + fgp);
                float inn = tanhf(inp);
                float tm1 = sigm(t1p + tanhf(pl[8 * H_ + c] * tval) + pl[9 * H_ + c]);
                float tm2 = sigm(t2p + tanhf(pl[10 * H_ + c] * tval) + pl[11 * H_ + c]);
                float fgcm = fg * cm;
                float igin = ig * inn;
                float cmh = fgcm + igin * tm1;
                cmreg[q] = fgcm + igin * tm2;
                float og = sigm(pl[5 * H_ + c] * cmh + ogp +
                                pl[7 * H_ + c] * tval + pl[6 * H_ + c]);
                float h = og * tanhf(cmh);
                hreg[q] = h;
                sA[r * H_ + c] = f2bf(h);   // next step's A matrix
            }
        }
        // no barrier needed: next iter only writes sX before the top barrier
    }

    #pragma unroll
    for (int q = 0; q < 16; ++q) {
        out[(size_t)(rowbase + r) * H_ + cb + q] = hreg[q];
        out[(size_t)B_ * H_ + (size_t)(rowbase + r) * H_ + cb + q] = cmreg[q];
    }
}

extern "C" void kernel_launch(void* const* d_in, const int* in_sizes, int n_in,
                              void* d_out, int out_size, void* d_ws, size_t ws_size,
                              hipStream_t stream) {
    const float* x       = (const float*)d_in[0];
    const float* t       = (const float*)d_in[1];
    const float* fg_w_c  = (const float*)d_in[2];
    const float* fg_w_h  = (const float*)d_in[3];
    const float* fg_w_x  = (const float*)d_in[4];
    const float* fg_b    = (const float*)d_in[5];
    const float* ig_w_c  = (const float*)d_in[6];
    const float* ig_w_h  = (const float*)d_in[7];
    const float* ig_w_x  = (const float*)d_in[8];
    const float* ig_b    = (const float*)d_in[9];
    const float* in_w_h  = (const float*)d_in[10];
    const float* in_w_x  = (const float*)d_in[11];
    const float* in_b    = (const float*)d_in[12];
    const float* og_w_cn = (const float*)d_in[13];
    const float* og_w_h  = (const float*)d_in[14];
    const float* og_w_x  = (const float*)d_in[15];
    const float* og_b    = (const float*)d_in[16];
    const float* og_w_t  = (const float*)d_in[17];
    const float* tg1_w_x = (const float*)d_in[18];
    const float* tg1_w_t = (const float*)d_in[19];
    const float* tg1_b   = (const float*)d_in[20];
    const float* tg2_w_x = (const float*)d_in[21];
    const float* tg2_w_t = (const float*)d_in[22];
    const float* tg2_b   = (const float*)d_in[23];

    // workspace: [0, 1MB) bf16 weights; [1MB, 1MB+201MB) precomputed x-side
    unsigned short* wh = (unsigned short*)d_ws;
    unsigned short* wx = wh + 4 * H_ * H_;
    unsigned short* pre = (unsigned short*)((char*)d_ws + (1u << 20));
    const size_t pre_bytes = (size_t)N_ * PRE_NSTRIDE * sizeof(unsigned short);
    const bool use_pre = ws_size >= (1u << 20) + pre_bytes;

    const int nh = H_ * H_, nx = H_ * X_;
    cvt_bf16<<<(nh + 255) / 256, 256, 0, stream>>>(ig_w_h, wh + 0 * nh, nh);
    cvt_bf16<<<(nh + 255) / 256, 256, 0, stream>>>(fg_w_h, wh + 1 * nh, nh);
    cvt_bf16<<<(nh + 255) / 256, 256, 0, stream>>>(in_w_h, wh + 2 * nh, nh);
    cvt_bf16<<<(nh + 255) / 256, 256, 0, stream>>>(og_w_h, wh + 3 * nh, nh);
    cvt_bf16<<<(nx + 255) / 256, 256, 0, stream>>>(ig_w_x, wx + 0 * nx, nx);
    cvt_bf16<<<(nx + 255) / 256, 256, 0, stream>>>(fg_w_x, wx + 1 * nx, nx);
    cvt_bf16<<<(nx + 255) / 256, 256, 0, stream>>>(in_w_x, wx + 2 * nx, nx);
    cvt_bf16<<<(nx + 255) / 256, 256, 0, stream>>>(og_w_x, wx + 3 * nx, nx);
    cvt_bf16<<<(nx + 255) / 256, 256, 0, stream>>>(tg1_w_x, wx + 4 * nx, nx);
    cvt_bf16<<<(nx + 255) / 256, 256, 0, stream>>>(tg2_w_x, wx + 5 * nx, nx);

    ColParams cp{ig_w_c, ig_b, fg_w_c, fg_b, in_b,
                 og_w_cn, og_b, og_w_t,
                 tg1_w_t, tg1_b, tg2_w_t, tg2_b};

    if (use_pre) {
        xgemm_pre<<<dim3(N_, B_ / 16), 256, 0, stream>>>(x, wx, pre);
        timelstm_scan<true><<<B_ / 16, 256, LDS_BYTES, stream>>>(
            x, t, wh, wx, pre, cp, (float*)d_out);
    } else {
        timelstm_scan<false><<<B_ / 16, 256, LDS_BYTES, stream>>>(
            x, t, wh, wx, nullptr, cp, (float*)d_out);
    }
}
